// HGT_39238821216885
// MI455X (gfx1250) — matrix operand order repaired and checked by reading, not compiled, not run on hardware
//
#include <hip/hip_runtime.h>
#include <hip/hip_bf16.h>

#define NN   50000
#define FD   256
#define CD   256
#define HH   8
#define DDIM 32
#define EE   200000
#define OUTD 16

typedef __attribute__((ext_vector_type(16))) __bf16 v16bf;
typedef __attribute__((ext_vector_type(8)))  float  v8f;

__device__ __forceinline__ float gelu_exact(float x) {
    return 0.5f * x * (1.0f + erff(x * 0.7071067811865476f));
}

// Monotone float <-> uint encoding so atomicMax(u32) gives float max.
__device__ __forceinline__ unsigned fenc(float f) {
    unsigned u = __float_as_uint(f);
    return (u & 0x80000000u) ? ~u : (u | 0x80000000u);
}
__device__ __forceinline__ float fdec(unsigned u) {
    return __uint_as_float((u & 0x80000000u) ? (u & 0x7FFFFFFFu) : ~u);
}
#define ENC_NEG_INF 0x007FFFFFu  // fenc(-inf)

// ---------------------------------------------------------------------------
// Fold per-head 32x32 relation transforms into K/V projection weights/biases.
// wck[c, h*32+e] = sum_d k_w0[c, h*32+d] * a_rel0[h,d,e]   (same for v/m_rel)
// ---------------------------------------------------------------------------
__global__ void combine_rel_kernel(const float* __restrict__ kw, const float* __restrict__ kb,
                                   const float* __restrict__ vw, const float* __restrict__ vb,
                                   const float* __restrict__ arel, const float* __restrict__ mrel,
                                   float* __restrict__ wck, float* __restrict__ bck,
                                   float* __restrict__ wcv, float* __restrict__ bcv) {
    int id = blockIdx.x * blockDim.x + threadIdx.x;
    if (id < CD * CD) {
        int c = id / CD, ce = id % CD;
        int h = ce / DDIM, e = ce % DDIM;
        float sk = 0.f, sv = 0.f;
        #pragma unroll 8
        for (int d = 0; d < DDIM; ++d) {
            float ar = arel[h * DDIM * DDIM + d * DDIM + e];
            float mr = mrel[h * DDIM * DDIM + d * DDIM + e];
            sk += kw[c * CD + h * DDIM + d] * ar;
            sv += vw[c * CD + h * DDIM + d] * mr;
        }
        wck[id] = sk; wcv[id] = sv;
    } else if (id < CD * CD + CD) {
        int ce = id - CD * CD;
        int h = ce / DDIM, e = ce % DDIM;
        float sk = 0.f, sv = 0.f;
        for (int d = 0; d < DDIM; ++d) {
            sk += kb[h * DDIM + d] * arel[h * DDIM * DDIM + d * DDIM + e];
            sv += vb[h * DDIM + d] * mrel[h * DDIM * DDIM + d * DDIM + e];
        }
        bck[ce] = sk; bcv[ce] = sv;
    }
}

// ---------------------------------------------------------------------------
// Pre-pack an fp32 [K, Nout] weight matrix into bf16 WMMA B-fragment order:
//   P[((nt*(K/32) + k32)*32 + lane)*16 + e] = W[k32*32 + (e&7) + (e>=8?16:0) + (lane>=16?8:0)][nt*16 + (lane&15)]
// so the GEMM loop loads one v16bf (32B/lane, wave-contiguous) per fragment.
// ---------------------------------------------------------------------------
__global__ void pack_weight_kernel(const float* __restrict__ W, __bf16* __restrict__ P,
                                   int K, int Nout) {
    int id = blockIdx.x * blockDim.x + threadIdx.x;
    if (id >= K * Nout) return;
    int e    = id & 15;
    int lane = (id >> 4) & 31;
    int k32  = (id >> 9) % (K >> 5);
    int nt   = id / (512 * (K >> 5));
    int k = k32 * 32 + (e & 7) + ((e >> 3) << 4) + ((lane >= 16) ? 8 : 0);
    int n = nt * 16 + (lane & 15);
    P[id] = (__bf16)W[(size_t)k * Nout + n];
}

// ---------------------------------------------------------------------------
// Fused GEMM: C[M, Nout] = epi( (GELU_A?gelu(A):A) @ W + bias )
//   NB 16-col tiles per wave (branch-free), K-loop step 32,
//   v_wmma_f32_16x16x32_bf16, pre-packed bf16 weights.
//   actOut==1 -> ReLU. mixY!=null -> C = beta*val + (1-beta)*mixY, beta=sigmoid(*skipPtr)
// ---------------------------------------------------------------------------
template <int NB, bool GELU_A>
__global__ void gemm_wmma_bf16(const float* __restrict__ A, const __bf16* __restrict__ Wp,
                               const float* __restrict__ bias, float* __restrict__ Cmat,
                               int M, int K, int Nout, int actOut,
                               const float* __restrict__ mixY, const float* __restrict__ skipPtr) {
    const int lane = threadIdx.x;             // wave32
    const int m0   = blockIdx.x * 16;
    const int ntb  = blockIdx.y * NB;         // first 16-col tile index
    const int kt   = K >> 5;                  // K/32 steps

    v8f acc[NB];
    #pragma unroll
    for (int t = 0; t < NB; ++t) acc[t] = (v8f){0.f, 0.f, 0.f, 0.f, 0.f, 0.f, 0.f, 0.f};

    const int mrow = m0 + (lane & 15);
    const int ahi  = (lane >= 16) ? 8 : 0;    // A-fragment K offset for upper half-wave
    const int ncol = lane & 15;
    const float* Arow = A + (size_t)mrow * K;

    for (int k32 = 0; k32 < kt; ++k32) {
        const int ko = (k32 << 5) + ahi;
        // A fragment 16x32 (ISA 16-bit A layout): two contiguous 8-float chunks/lane
        float4 a0 = *(const float4*)(Arow + ko);
        float4 a1 = *(const float4*)(Arow + ko + 4);
        float4 a2 = *(const float4*)(Arow + ko + 16);
        float4 a3 = *(const float4*)(Arow + ko + 20);
        if (GELU_A) {
            a0.x = gelu_exact(a0.x); a0.y = gelu_exact(a0.y); a0.z = gelu_exact(a0.z); a0.w = gelu_exact(a0.w);
            a1.x = gelu_exact(a1.x); a1.y = gelu_exact(a1.y); a1.z = gelu_exact(a1.z); a1.w = gelu_exact(a1.w);
            a2.x = gelu_exact(a2.x); a2.y = gelu_exact(a2.y); a2.z = gelu_exact(a2.z); a2.w = gelu_exact(a2.w);
            a3.x = gelu_exact(a3.x); a3.y = gelu_exact(a3.y); a3.z = gelu_exact(a3.z); a3.w = gelu_exact(a3.w);
        }
        v16bf afrag;
        afrag[0]  = (__bf16)a0.x; afrag[1]  = (__bf16)a0.y; afrag[2]  = (__bf16)a0.z; afrag[3]  = (__bf16)a0.w;
        afrag[4]  = (__bf16)a1.x; afrag[5]  = (__bf16)a1.y; afrag[6]  = (__bf16)a1.z; afrag[7]  = (__bf16)a1.w;
        afrag[8]  = (__bf16)a2.x; afrag[9]  = (__bf16)a2.y; afrag[10] = (__bf16)a2.z; afrag[11] = (__bf16)a2.w;
        afrag[12] = (__bf16)a3.x; afrag[13] = (__bf16)a3.y; afrag[14] = (__bf16)a3.z; afrag[15] = (__bf16)a3.w;

        #pragma unroll
        for (int t = 0; t < NB; ++t) {
            const v16bf bfrag = *(const v16bf*)(Wp + ((size_t)((ntb + t) * kt + k32) * 32 + lane) * 16);
            acc[t] = __builtin_amdgcn_wmma_f32_16x16x32_bf16(
                false, afrag, false, bfrag, (short)0, acc[t], false, false);
        }
    }

    float beta = 1.f, ombeta = 0.f;
    if (mixY) { float s = *skipPtr; beta = 1.f / (1.f + expf(-s)); ombeta = 1.f - beta; }

    #pragma unroll
    for (int t = 0; t < NB; ++t) {
        int n = (ntb + t) * 16 + ncol;
        float bv = bias[n];
        #pragma unroll
        for (int r = 0; r < 8; ++r) {
            int m = m0 + r + ahi;                 // C/D layout: M=r (+8 for upper half-wave)
            float val = acc[t][r] + bv;
            if (actOut) val = fmaxf(val, 0.f);
            if (mixY) val = beta * val + ombeta * mixY[(size_t)m * Nout + n];
            Cmat[(size_t)m * Nout + n] = val;
        }
    }
}

// ---------------------------------------------------------------------------
// Workspace init: amax=enc(-inf), den=0, agg=0
// ---------------------------------------------------------------------------
__global__ void init_ws_kernel(unsigned* __restrict__ amax, float* __restrict__ den,
                               float* __restrict__ agg) {
    int id = blockIdx.x * blockDim.x + threadIdx.x;
    if (id < NN * HH) { amax[id] = ENC_NEG_INF; den[id] = 0.f; }
    if (id < NN * CD) { agg[id] = 0.f; }
}

// Pass 1: attention logits + segment max (edge type 0 only: src type0 -> dst type1)
__global__ void edge_logits_kernel(const long long* __restrict__ ei,
                                   const float* __restrict__ q1, const float* __restrict__ krel,
                                   const float* __restrict__ prel,
                                   float* __restrict__ logits, unsigned* __restrict__ amax) {
    int id = blockIdx.x * blockDim.x + threadIdx.x;
    if (id >= EE * HH) return;
    int e = id / HH, h = id % HH;
    int src = (int)ei[e];
    int dst = (int)ei[EE + e];
    const float4* qp = (const float4*)(q1 + (size_t)dst * CD + h * DDIM);
    const float4* kp = (const float4*)(krel + (size_t)src * CD + h * DDIM);
    float s = 0.f;
    #pragma unroll
    for (int i = 0; i < 8; ++i) {
        float4 a = qp[i], b = kp[i];
        s += a.x * b.x + a.y * b.y + a.z * b.z + a.w * b.w;
    }
    s *= prel[h] * 0.17677669529663687f;   // p_rel[0,h] / sqrt(32)
    logits[id] = s;
    atomicMax(&amax[dst * HH + h], fenc(s));
}

// Pass 2: ex = exp(logit - amax); segment-sum denominator
__global__ void edge_exp_kernel(const long long* __restrict__ ei,
                                float* __restrict__ logits, const unsigned* __restrict__ amax,
                                float* __restrict__ den) {
    int id = blockIdx.x * blockDim.x + threadIdx.x;
    if (id >= EE * HH) return;
    int e = id / HH, h = id % HH;
    int dst = (int)ei[EE + e];
    float am = fdec(amax[dst * HH + h]);
    if (!isfinite(am)) am = 0.f;
    float ex = expf(logits[id] - am);
    logits[id] = ex;
    atomicAdd(&den[dst * HH + h], ex);
}

// Pass 3: agg[dst] += v_rel[src] * (ex / (den+1e-16))
__global__ void edge_agg_kernel(const long long* __restrict__ ei,
                                const float* __restrict__ logits, const float* __restrict__ den,
                                const float* __restrict__ vrel, float* __restrict__ agg) {
    int id = blockIdx.x * blockDim.x + threadIdx.x;
    if (id >= EE * HH) return;
    int e = id / HH, h = id % HH;
    int src = (int)ei[e];
    int dst = (int)ei[EE + e];
    float w = logits[id] / (den[dst * HH + h] + 1e-16f);
    const float4* vp = (const float4*)(vrel + (size_t)src * CD + h * DDIM);
    float* ap = agg + (size_t)dst * CD + h * DDIM;
    #pragma unroll
    for (int i = 0; i < 8; ++i) {
        float4 v = vp[i];
        atomicAdd(ap + 4 * i + 0, v.x * w);
        atomicAdd(ap + 4 * i + 1, v.y * w);
        atomicAdd(ap + 4 * i + 2, v.z * w);
        atomicAdd(ap + 4 * i + 3, v.w * w);
    }
}

// ---------------------------------------------------------------------------
extern "C" void kernel_launch(void* const* d_in, const int* in_sizes, int n_in,
                              void* d_out, int out_size, void* d_ws, size_t ws_size,
                              hipStream_t stream) {
    (void)in_sizes; (void)n_in; (void)out_size; (void)ws_size;

    const float*     x      = (const float*)d_in[0];
    const long long* ei     = (const long long*)d_in[1];   // int64 [3,2,E]; type 0 at base
    const float*     lin_w  = (const float*)d_in[2];
    const float*     lin_b  = (const float*)d_in[3];
    const float*     k_w    = (const float*)d_in[4];
    const float*     k_b    = (const float*)d_in[5];
    const float*     q_w    = (const float*)d_in[6];
    const float*     q_b    = (const float*)d_in[7];
    const float*     v_w    = (const float*)d_in[8];
    const float*     v_b    = (const float*)d_in[9];
    const float*     a_w    = (const float*)d_in[10];
    const float*     a_b    = (const float*)d_in[11];
    const float*     skip   = (const float*)d_in[12];
    const float*     a_rel  = (const float*)d_in[13];
    const float*     m_rel  = (const float*)d_in[14];
    const float*     p_rel  = (const float*)d_in[15];
    const float*     out_w  = (const float*)d_in[16];
    const float*     out_b  = (const float*)d_in[17];
    float*           out    = (float*)d_out;

    // Dead-code-eliminated slices: only node-type-0 K/V, node-type-1 Q,
    // edge type 0, and the labeled ('item' = index 1) output path are live.
    const float* x0     = x;
    const float* x1     = x + (size_t)NN * FD;
    const float* lin_w0 = lin_w;
    const float* lin_w1 = lin_w + (size_t)FD * CD;
    const float* lin_b0 = lin_b;
    const float* lin_b1 = lin_b + CD;
    const float* q_w1   = q_w + (size_t)CD * CD;
    const float* q_b1   = q_b + CD;
    const float* a_w1   = a_w + (size_t)CD * CD;
    const float* a_b1   = a_b + CD;
    const float* skip1  = skip + 1;

    // Workspace layout (float units; every region a multiple of 64 floats)
    float* ws = (float*)d_ws;
    const size_t NC = (size_t)NN * CD;
    float* h0     = ws;                 // later reused as 'o'
    float* h1     = h0 + NC;
    float* krel   = h1 + NC;
    float* vrel   = krel + NC;
    float* q1     = vrel + NC;
    float* agg    = q1 + NC;
    float* wck    = agg + NC;                         // fp32 combined K weights
    float* wcv    = wck + (size_t)CD * CD;
    float* bck    = wcv + (size_t)CD * CD;
    float* bcv    = bck + CD;
    float* logits = bcv + CD;                         // E*H
    unsigned* amax = (unsigned*)(logits + (size_t)EE * HH);
    float* den    = (float*)(amax + (size_t)NN * HH);
    // packed bf16 weights (each 256x256 = 64K bf16 = 32K float slots)
    __bf16* pLin0 = (__bf16*)(den + (size_t)NN * HH);
    __bf16* pLin1 = pLin0 + (size_t)FD * CD;
    __bf16* pQ    = pLin1 + (size_t)FD * CD;
    __bf16* pA    = pQ    + (size_t)CD * CD;
    __bf16* pK    = pA    + (size_t)CD * CD;
    __bf16* pV    = pK    + (size_t)CD * CD;
    __bf16* pOut  = pV    + (size_t)CD * CD;          // 256x16

    const int PB = 256;
    const int wElems = CD * CD;

    // 1. Fold relation transforms into combined projection weights
    combine_rel_kernel<<<(CD * CD + CD + 255) / 256, 256, 0, stream>>>(
        k_w, k_b, v_w, v_b, a_rel, m_rel, wck, bck, wcv, bcv);

    // 2. Pre-pack all weight matrices into WMMA B-fragment bf16 order
    pack_weight_kernel<<<(wElems + PB - 1) / PB, PB, 0, stream>>>(lin_w0, pLin0, FD, CD);
    pack_weight_kernel<<<(wElems + PB - 1) / PB, PB, 0, stream>>>(lin_w1, pLin1, FD, CD);
    pack_weight_kernel<<<(wElems + PB - 1) / PB, PB, 0, stream>>>(q_w1,   pQ,    CD, CD);
    pack_weight_kernel<<<(wElems + PB - 1) / PB, PB, 0, stream>>>(a_w1,   pA,    CD, CD);
    pack_weight_kernel<<<(wElems + PB - 1) / PB, PB, 0, stream>>>(wck,    pK,    CD, CD);
    pack_weight_kernel<<<(wElems + PB - 1) / PB, PB, 0, stream>>>(wcv,    pV,    CD, CD);
    pack_weight_kernel<<<(CD * OUTD + PB - 1) / PB, PB, 0, stream>>>(out_w, pOut, CD, OUTD);

    // 3. Init softmax/agg workspace
    init_ws_kernel<<<((int)NC + 255) / 256, 256, 0, stream>>>(amax, den, agg);

    // 4. h = relu(x @ lin_w + lin_b) for both node types
    dim3 gFull(NN / 16, CD / 64);
    gemm_wmma_bf16<4, false><<<gFull, 32, 0, stream>>>(x0, pLin0, lin_b0, h0, NN, FD, CD, 1, nullptr, nullptr);
    gemm_wmma_bf16<4, false><<<gFull, 32, 0, stream>>>(x1, pLin1, lin_b1, h1, NN, FD, CD, 1, nullptr, nullptr);

    // 5. Projections (relation transform pre-folded into pK/pV)
    gemm_wmma_bf16<4, false><<<gFull, 32, 0, stream>>>(h0, pK, bck,  krel, NN, CD, CD, 0, nullptr, nullptr);
    gemm_wmma_bf16<4, false><<<gFull, 32, 0, stream>>>(h0, pV, bcv,  vrel, NN, CD, CD, 0, nullptr, nullptr);
    gemm_wmma_bf16<4, false><<<gFull, 32, 0, stream>>>(h1, pQ, q_b1, q1,   NN, CD, CD, 0, nullptr, nullptr);

    // 6. Edge-type-0 attention: logits -> segment softmax -> scatter-add
    int eg = (EE * HH + 255) / 256;
    edge_logits_kernel<<<eg, 256, 0, stream>>>(ei, q1, krel, p_rel, logits, amax);
    edge_exp_kernel<<<eg, 256, 0, stream>>>(ei, logits, amax, den);
    edge_agg_kernel<<<eg, 256, 0, stream>>>(ei, logits, den, vrel, agg);

    // 7. o = sigmoid(skip1)*(gelu(agg) @ a_w1 + a_b1) + (1-sigmoid)*h1   (into h0)
    gemm_wmma_bf16<4, true><<<gFull, 32, 0, stream>>>(agg, pA, a_b1, h0, NN, CD, CD, 0, h1, skip1);

    // 8. out = o @ out_w + out_b    [N,16]
    dim3 gOut(NN / 16, 1);
    gemm_wmma_bf16<1, false><<<gOut, 32, 0, stream>>>(h0, pOut, out_b, out, NN, CD, OUTD, 0, nullptr, nullptr);
}
